// ContrastiveLoss_13718125543815
// MI455X (gfx1250) — matrix operand order repaired and checked by reading, not compiled, not run on hardware
//
#include <hip/hip_runtime.h>
#include <hip/hip_bf16.h>

typedef __attribute__((ext_vector_type(16))) _Float16 v16h;
typedef __attribute__((ext_vector_type(8)))  float    v8f;

#define TEMP_INV 10.0f   // 1 / TEMPERATURE

// ---------------------------------------------------------------------------
// Kernel 1: row L2-normalize fp32 -> fp16.  One wave32 per row (D=128 -> each
// lane handles 4 contiguous floats), wave reduction via shfl_xor.
// ---------------------------------------------------------------------------
__global__ __launch_bounds__(256)
void normalize_rows_f16(const float* __restrict__ f, _Float16* __restrict__ g,
                        int B, int D) {
    const int wave = (blockIdx.x * blockDim.x + threadIdx.x) >> 5;
    const int lane = threadIdx.x & 31;
    const float* src = f + (size_t)wave * D;
    float ss = 0.0f;
    for (int d = lane * 4; d < D; d += 128) {
        float4 v = *(const float4*)(src + d);
        ss += v.x * v.x + v.y * v.y + v.z * v.z + v.w * v.w;
    }
#pragma unroll
    for (int m = 1; m < 32; m <<= 1) ss += __shfl_xor(ss, m, 32);
    const float scale = rsqrtf(fmaxf(ss, 1e-16f));
    _Float16* dst = g + (size_t)wave * D;
    for (int d = lane * 4; d < D; d += 128) {
        float4 v = *(const float4*)(src + d);
        union { _Float16 h[4]; float2 f2; } u;
        u.h[0] = (_Float16)(v.x * scale);
        u.h[1] = (_Float16)(v.y * scale);
        u.h[2] = (_Float16)(v.z * scale);
        u.h[3] = (_Float16)(v.w * scale);
        *(float2*)(dst + d) = u.f2;
    }
}

// ---------------------------------------------------------------------------
// Kernel 2: gather labels, build 2-bit class per row, 4-bin histogram.
// pos class of row = cls^1 (same t, diff p); neg class = cls^2.
// ---------------------------------------------------------------------------
__global__ void class_hist(const int* __restrict__ data_ix,
                           const int* __restrict__ tt, const int* __restrict__ tp,
                           int* __restrict__ cls, int* __restrict__ hist, int B) {
    const int i = blockIdx.x * blockDim.x + threadIdx.x;
    if (i >= B) return;
    const int ix = data_ix[i];
    const int c = (tt[ix] << 1) | tp[ix];
    cls[i] = c;
    atomicAdd(&hist[c], 1);
}

// ---------------------------------------------------------------------------
// Kernel 3: fused  G*G^T  +  contrastive epilogue.
// One wave per 16-row stripe; sweep all column tiles with
// v_wmma_f32_16x16x32_f16 (D=128 -> 4 chained WMMAs per 16x16 tile).
// Fragment layouts follow CDNA5 ISA 7.12.2:
//   A 16x32 f16: lane(0-15)=M, half h: VGPR0-3 <- K=8h+0..7, VGPR4-7 <- K=16+8h+0..7
//   B 32x16 f16: lane(0-15)=N, half h: VGPR0-3 <- K=8h+0..7, VGPR4-7 <- K=16+8h+0..7
//   C 16x16 f32: VGPR r, lane l: row = r + 8*(l>=16), col = l&15
// NOTE: no early-exit guards -> EXEC is all-1s at every WMMA (ISA requirement).
// ---------------------------------------------------------------------------
__global__ __launch_bounds__(256)
void sim_tiles_wmma(const _Float16* __restrict__ g, const int* __restrict__ cls,
                    float* __restrict__ rowPosExp, float* __restrict__ rowNegExp,
                    float* __restrict__ rowPosLog, int B, int D) {
    const int lane    = threadIdx.x & 31;
    const int waveId  = threadIdx.x >> 5;
    const int rowTile = blockIdx.x * (blockDim.x >> 5) + waveId;
    const int r0      = rowTile << 4;
    const int half    = lane >> 4;
    const int l       = lane & 15;

    // A fragments for this row stripe: resident across the whole column sweep.
    v16h a[4];
    {
        const _Float16* ap = g + (size_t)(r0 + l) * D;
#pragma unroll
        for (int kb = 0; kb < 4; ++kb) {
            const int k0 = kb * 32 + half * 8;
            ((float4*)&a[kb])[0] = *(const float4*)(ap + k0);        // K = k0..k0+7
            ((float4*)&a[kb])[1] = *(const float4*)(ap + k0 + 16);   // K = k0+16..k0+23
        }
    }

    // Per-accumulator-slot row classes (row = r0 + r + 8*half).
    int rcp[8], rcn[8];
#pragma unroll
    for (int r = 0; r < 8; ++r) {
        const int rc = cls[r0 + r + 8 * half];
        rcp[r] = rc ^ 1;   // positive column class
        rcn[r] = rc ^ 2;   // negative column class
    }

    float posE[8], negE[8], posL[8];
#pragma unroll
    for (int r = 0; r < 8; ++r) { posE[r] = 0.f; negE[r] = 0.f; posL[r] = 0.f; }

    const int nColTiles = B >> 4;
    for (int ct = 0; ct < nColTiles; ++ct) {
        const int c0 = ct << 4;
        const _Float16* bp = g + (size_t)(c0 + l) * D + half * 8;
        v16h b[4];
#pragma unroll
        for (int kb = 0; kb < 4; ++kb) {
            ((float4*)&b[kb])[0] = *(const float4*)(bp + kb * 32);       // K = 32kb+8h+0..7
            ((float4*)&b[kb])[1] = *(const float4*)(bp + kb * 32 + 16);  // K = 32kb+16+8h+0..7
        }
        const int cj = cls[c0 + l];
        if (ct + 1 < nColTiles)   // pull next column stripe toward L0/L2
            __builtin_prefetch(g + (size_t)(c0 + 16 + l) * D + half * 16, 0, 0);

        v8f c = {};
#pragma unroll
        for (int kb = 0; kb < 4; ++kb)
            c = __builtin_amdgcn_wmma_f32_16x16x32_f16(false, a[kb], false, b[kb],
                                                       (short)0, c, false, false);

        // Fused epilogue: logits = sim/T, masked exp/logit accumulation.
#pragma unroll
        for (int r = 0; r < 8; ++r) {
            const float logit = c[r] * TEMP_INV;
            const float e = __expf(logit);
            const bool isP = (cj == rcp[r]);
            const bool isN = (cj == rcn[r]);
            posE[r] += isP ? e : 0.0f;
            posL[r] += isP ? logit : 0.0f;
            negE[r] += isN ? e : 0.0f;
        }
    }

    // Reduce over the 16 column-lanes of each half; lane l==0 of each half
    // writes the 3 per-row scalars (this wave owns these rows -> no atomics).
#pragma unroll
    for (int r = 0; r < 8; ++r) {
#pragma unroll
        for (int m = 1; m < 16; m <<= 1) {
            posE[r] += __shfl_xor(posE[r], m, 32);
            negE[r] += __shfl_xor(negE[r], m, 32);
            posL[r] += __shfl_xor(posL[r], m, 32);
        }
        if (l == 0) {
            const int row = r0 + r + 8 * half;
            rowPosExp[row] = posE[r];
            rowNegExp[row] = negE[r];
            rowPosLog[row] = posL[r];
        }
    }
}

// ---------------------------------------------------------------------------
// Kernel 4: finalize scalar loss.
// log_probs_sum(i) = posLog(i) - pos_cnt(i) * log(posExp(i) + negExp(i))
// ---------------------------------------------------------------------------
__global__ void finalize_loss(const int* __restrict__ cls, const int* __restrict__ hist,
                              const float* __restrict__ rowPosExp,
                              const float* __restrict__ rowNegExp,
                              const float* __restrict__ rowPosLog,
                              float* __restrict__ out, int B) {
    __shared__ float sacc[256];
    __shared__ int shist[4];
    if (threadIdx.x < 4) shist[threadIdx.x] = hist[threadIdx.x];
    __syncthreads();
    float acc = 0.0f;
    for (int i = threadIdx.x; i < B; i += blockDim.x) {
        const int rc = cls[i];
        const int pc = shist[rc ^ 1];
        const int nc = shist[rc ^ 2];
        if (pc > 0 && nc > 0) {
            const float denom = logf(rowPosExp[i] + rowNegExp[i]);
            const float lps = rowPosLog[i] - (float)pc * denom;
            acc += lps / (float)pc;
        }
    }
    sacc[threadIdx.x] = acc;
    __syncthreads();
    for (int s = blockDim.x >> 1; s > 0; s >>= 1) {
        if (threadIdx.x < s) sacc[threadIdx.x] += sacc[threadIdx.x + s];
        __syncthreads();
    }
    if (threadIdx.x == 0) out[0] = -sacc[0] / (float)B;
}

// ---------------------------------------------------------------------------
extern "C" void kernel_launch(void* const* d_in, const int* in_sizes, int n_in,
                              void* d_out, int out_size, void* d_ws, size_t ws_size,
                              hipStream_t stream) {
    const float* features  = (const float*)d_in[0];
    const int*   data_ix   = (const int*)d_in[1];
    const int*   targets_t = (const int*)d_in[2];
    const int*   targets_p = (const int*)d_in[3];

    const int B = in_sizes[1];          // 8192
    const int D = in_sizes[0] / B;      // 128

    // Workspace carve-out (256B-aligned slabs).
    char* ws = (char*)d_ws;
    size_t off = 0;
    auto carve = [&](size_t bytes) -> void* {
        void* p = ws + off;
        off = (off + bytes + 255) & ~(size_t)255;
        return p;
    };
    _Float16* g    = (_Float16*)carve((size_t)B * D * sizeof(_Float16)); // 2 MB
    int*      cls  = (int*)carve((size_t)B * sizeof(int));
    int*      hist = (int*)carve(4 * sizeof(int));
    float*    rPE  = (float*)carve((size_t)B * sizeof(float));
    float*    rNE  = (float*)carve((size_t)B * sizeof(float));
    float*    rPL  = (float*)carve((size_t)B * sizeof(float));

    hipMemsetAsync(hist, 0, 4 * sizeof(int), stream);

    // 1) normalize: one wave per row, 8 waves / block.
    normalize_rows_f16<<<B / 8, 256, 0, stream>>>(features, g, B, D);

    // 2) classes + histogram.
    class_hist<<<(B + 255) / 256, 256, 0, stream>>>(data_ix, targets_t, targets_p,
                                                    cls, hist, B);

    // 3) fused WMMA similarity + epilogue: 512 row stripes, 8 waves / block.
    sim_tiles_wmma<<<(B / 16) / 8, 256, 0, stream>>>(g, cls, rPE, rNE, rPL, B, D);

    // 4) scalar loss.
    finalize_loss<<<1, 256, 0, stream>>>(cls, hist, rPE, rNE, rPL, (float*)d_out, B);
}
